// KNLTransformerEncoder_63093069578867
// MI455X (gfx1250) — matrix operand order repaired and checked by reading, not compiled, not run on hardware
//
#include <hip/hip_runtime.h>

// ---------------- problem constants ----------------
#define LNUM 6
#define DMODEL 1024
#define HEADS 16
#define FFDIM 4096
#define SEQ 1024
#define BATCH 8
#define DHEAD 64
#define TOK (BATCH * SEQ)   // 8192
#define CHUNK 8             // (b,h) pairs per attention pass; 128/CHUNK = 16 passes

// ---------------- WMMA types ----------------
typedef __attribute__((ext_vector_type(16))) __bf16 v16bf;
typedef __attribute__((ext_vector_type(8)))  float  v8f;

union FragBF { v16bf v; unsigned int u[8]; };

// ============================================================
// Generic batched bf16 GEMM:  C[z] = A[z](MxK) * B[z](KxN) (+bias)(+resid)(relu)
// A: bf16 row-major (lda). B: bf16; transB=1 -> [N][K] row-major (contig-K,
// the fast path: pre-transposed weights / K^T); transB=0 -> [K][N] (ctx GEMM).
// C: f32 or bf16 (cbf). Per-operand batch offset:
//    off = ((zb+z)/cnt)*so + ((zb+z)%cnt)*si
// Block tile 256(M) x TNB(N), K-step 32, 256 thr = 8 waves; wave w owns rows
// [w*32, w*32+32) (two 16-row fragments) and all TNB cols ->
// 2*TNB/16 WMMAs per K-step per wave, each B fragment reused twice.
// ============================================================
template <int TNB>
__global__ __launch_bounds__(256) void KNL_gemm_bf16(
    const __bf16* __restrict__ A,  long lda, long zbA, long cntA, long soA, long siA,
    const __bf16* __restrict__ Bm, long ldb, long zbB, long cntB, long soB, long siB, int transB,
    void* __restrict__ Cp, int cbf,  long ldc, long zbC, long cntC, long soC, long siC,
    const float* __restrict__ bias, const float* __restrict__ resid, int relu, int K)
{
    constexpr int NF = TNB / 16;
    __shared__ alignas(16) __bf16 sA[256 * 32];   // [m][k]
    __shared__ alignas(16) __bf16 sB[TNB * 32];   // [n][k] (K-contiguous)

    const int tid  = threadIdx.x;
    const int lane = tid & 31;
    const int wm   = tid >> 5;
    const int hi   = lane >> 4;

    const long z    = blockIdx.z;
    const long offA = ((zbA + z) / cntA) * soA + ((zbA + z) % cntA) * siA;
    const long offB = ((zbB + z) / cntB) * soB + ((zbB + z) % cntB) * siB;
    const long offC = ((zbC + z) / cntC) * soC + ((zbC + z) % cntC) * siC;

    const long mB = (long)blockIdx.y * 256;
    const long n0 = (long)blockIdx.x * TNB;

    v8f acc[2][NF];
#pragma unroll
    for (int h2 = 0; h2 < 2; ++h2)
#pragma unroll
        for (int nf = 0; nf < NF; ++nf)
#pragma unroll
            for (int i = 0; i < 8; ++i) acc[h2][nf][i] = 0.0f;

    const unsigned int* sA32 = (const unsigned int*)sA;
    const unsigned int* sB32 = (const unsigned int*)sB;

    for (int k0 = 0; k0 < K; k0 += 32) {
        // ---- stage A tile 256x32 bf16: pure copy, 4x b128 per thread ----
#pragma unroll
        for (int i = 0; i < 4; ++i) {
            int lin = (tid + i * 256) * 8;
            int r   = lin >> 5;
            int kk  = lin & 31;
            const __bf16* src = A + offA + (mB + r) * lda + (k0 + kk);
            *(uint4*)&sA[r * 32 + kk] = *(const uint4*)src;
            if (i == 0) __builtin_prefetch(src + 32, 0, 1);   // next K-slab
        }
        // ---- stage B tile TNBx32 (as [n][k]) ----
        if (transB) {
#pragma unroll
            for (int i = 0; i < TNB / 64; ++i) {
                int lin = (tid + i * 256) * 8;
                int n   = lin >> 5;
                int kk  = lin & 31;
                const __bf16* src = Bm + offB + (n0 + n) * ldb + (k0 + kk);
                *(uint4*)&sB[n * 32 + kk] = *(const uint4*)src;
                if (i == 0) __builtin_prefetch(src + 32, 0, 1);
            }
        } else {
#pragma unroll
            for (int i = 0; i < TNB / 64; ++i) {
                int lin = (tid + i * 256) * 8;
                int kk  = lin / TNB;
                int n   = lin % TNB;
                const __bf16* src = Bm + offB + (k0 + kk) * ldb + (n0 + n);
                uint4 vv = *(const uint4*)src;
                const __bf16* e = (const __bf16*)&vv;
#pragma unroll
                for (int j = 0; j < 8; ++j) sB[(n + j) * 32 + kk] = e[j];
                if (i == 0) __builtin_prefetch(src + 32 * ldb, 0, 1);
            }
        }
        __syncthreads();

        // ---- two A fragments (ISA 7.12.2 16-bit 16x32 layout) ----
        FragBF a0, a1;
        {
            int base = (wm * 32 + (lane & 15)) * 16 + hi * 4;  // uint index
#pragma unroll
            for (int p = 0; p < 4; ++p) {
                a0.u[p]     = sA32[base + p];
                a0.u[4 + p] = sA32[base + 8 + p];
                a1.u[p]     = sA32[base + 256 + p];       // +16 rows
                a1.u[4 + p] = sA32[base + 256 + 8 + p];
            }
        }
        // ---- B fragments + WMMA (each B reused for both A frags) ----
#pragma unroll
        for (int nf = 0; nf < NF; ++nf) {
            FragBF b;
            int bbase = (nf * 16 + (lane & 15)) * 16 + hi * 4;
#pragma unroll
            for (int p = 0; p < 8; ++p) b.u[p] = sB32[bbase + (p & 3) + (p >> 2) * 8];
            acc[0][nf] = __builtin_amdgcn_wmma_f32_16x16x32_bf16(
                false, a0.v, false, b.v, (short)0, acc[0][nf], false, false);
            acc[1][nf] = __builtin_amdgcn_wmma_f32_16x16x32_bf16(
                false, a1.v, false, b.v, (short)0, acc[1][nf], false, false);
        }
        __syncthreads();
    }

    // ---- epilogue (C/D layout: VGPR rr -> M=rr (+8 for hi lanes)) ----
    const int colLane = lane & 15;
    float*  Cf = (float*)Cp;
    __bf16* Cb = (__bf16*)Cp;
#pragma unroll
    for (int h2 = 0; h2 < 2; ++h2) {
        const int rowLocal = wm * 32 + h2 * 16 + hi * 8;
#pragma unroll
        for (int nf = 0; nf < NF; ++nf) {
            long col = n0 + nf * 16 + colLane;
            float bi = bias ? bias[col] : 0.0f;
#pragma unroll
            for (int rr = 0; rr < 8; ++rr) {
                long idx = offC + (mB + rowLocal + rr) * ldc + col;
                float v  = acc[h2][nf][rr] + bi;
                if (resid) v += resid[idx];
                if (relu)  v = fmaxf(v, 0.0f);
                if (cbf) Cb[idx] = (__bf16)v;
                else     Cf[idx] = v;
            }
        }
    }
}

// ============================================================
// Transpose + f32->bf16 convert: in[K][N] f32 -> out[N][K] bf16
// ============================================================
__global__ __launch_bounds__(256) void KNL_transpose_cvt(
    const float* __restrict__ in, __bf16* __restrict__ out, int K, int N)
{
    __shared__ float t[32][33];
    const int n0 = blockIdx.x * 32, k0 = blockIdx.y * 32;
    const int lx = threadIdx.x & 31, ly = threadIdx.x >> 5;
#pragma unroll
    for (int i = 0; i < 4; ++i)
        t[ly + 8 * i][lx] = in[(long)(k0 + ly + 8 * i) * N + (n0 + lx)];
    __syncthreads();
#pragma unroll
    for (int i = 0; i < 4; ++i)
        out[(long)(n0 + ly + 8 * i) * K + (k0 + lx)] = (__bf16)t[lx][ly + 8 * i];
}

// ============================================================
// LayerNorm: one block per row of 1024; writes f32 or bf16
// ============================================================
__global__ __launch_bounds__(256) void KNL_layernorm(
    const float* __restrict__ x, const float* __restrict__ g,
    const float* __restrict__ b, void* __restrict__ y, int obf)
{
    __shared__ float red[256];
    const long row = blockIdx.x;
    const int  tid = threadIdx.x;
    const float* xr = x + row * (long)DMODEL;

    float v0[4];
    float s = 0.0f;
#pragma unroll
    for (int t = 0; t < 4; ++t) { v0[t] = xr[tid + t * 256]; s += v0[t]; }
    red[tid] = s; __syncthreads();
    for (int o = 128; o > 0; o >>= 1) { if (tid < o) red[tid] += red[tid + o]; __syncthreads(); }
    const float mean = red[0] * (1.0f / DMODEL);
    __syncthreads();

    float vs = 0.0f;
#pragma unroll
    for (int t = 0; t < 4; ++t) { float d = v0[t] - mean; vs += d * d; }
    red[tid] = vs; __syncthreads();
    for (int o = 128; o > 0; o >>= 1) { if (tid < o) red[tid] += red[tid + o]; __syncthreads(); }
    const float inv = rsqrtf(red[0] * (1.0f / DMODEL) + 1e-6f);

#pragma unroll
    for (int t = 0; t < 4; ++t) {
        int j = tid + t * 256;
        float v = (v0[t] - mean) * inv * g[j] + b[j];
        if (obf) ((__bf16*)y)[row * (long)DMODEL + j] = (__bf16)v;
        else     ((float*)y)[row * (long)DMODEL + j]  = v;
    }
}

// ============================================================
// Masked, scaled softmax: reads f32 scores, writes bf16 probs.
// grid = (SEQ, CHUNK); global (b,h) index = z0 + blockIdx.y
// ============================================================
__global__ __launch_bounds__(256) void KNL_softmax(
    const float* __restrict__ scf, __bf16* __restrict__ scb,
    const int* __restrict__ lengths, int z0)
{
    __shared__ float red[256];
    const float scale = 0.125f;  // 1/sqrt(64)
    const int z   = blockIdx.y;
    const int row = blockIdx.x;
    const int b   = (z0 + z) >> 4;  // / HEADS
    const int len = lengths[b];
    const long base = ((long)z << 20) + ((long)row << 10);
    const float* p = scf + base;
    __bf16*      q = scb + base;
    const int tid = threadIdx.x;

    float vals[4];
    float mx = -3.0e38f;
#pragma unroll
    for (int t = 0; t < 4; ++t) {
        int j = tid + t * 256;
        float v = (j < len) ? p[j] * scale : -3.0e38f;
        vals[t] = v; mx = fmaxf(mx, v);
    }
    red[tid] = mx; __syncthreads();
    for (int o = 128; o > 0; o >>= 1) { if (tid < o) red[tid] = fmaxf(red[tid], red[tid + o]); __syncthreads(); }
    mx = red[0]; __syncthreads();

    float e[4];
    float sum = 0.0f;
#pragma unroll
    for (int t = 0; t < 4; ++t) {
        int j = tid + t * 256;
        float ev = (j < len) ? __expf(vals[t] - mx) : 0.0f;
        e[t] = ev; sum += ev;
    }
    red[tid] = sum; __syncthreads();
    for (int o = 128; o > 0; o >>= 1) { if (tid < o) red[tid] += red[tid + o]; __syncthreads(); }
    const float inv = 1.0f / red[0];
#pragma unroll
    for (int t = 0; t < 4; ++t) { int j = tid + t * 256; q[j] = (__bf16)(e[t] * inv); }
}

// ============================================================
// Embedding gather (f32 residual stream)
// ============================================================
__global__ __launch_bounds__(256) void KNL_embed(
    const int* __restrict__ src, const float* __restrict__ emb, float* __restrict__ out)
{
    const long t   = blockIdx.x;
    const long tok = src[t];
    const float4* e = (const float4*)(emb + tok * (long)DMODEL);
    float4*       o = (float4*)(out + t * (long)DMODEL);
    o[threadIdx.x] = e[threadIdx.x];
}

// ============================================================
// host helpers
// ============================================================
static void launch_gemm(hipStream_t st, int TN,
    const __bf16* A,  long lda, long zbA, long cntA, long soA, long siA,
    const __bf16* Bm, long ldb, long zbB, long cntB, long soB, long siB, int transB,
    void* C, int cbf,  long ldc, long zbC, long cntC, long soC, long siC,
    const float* bias, const float* resid, int relu,
    int M, int N, int K, int batch)
{
    dim3 grid(N / TN, M / 256, batch);
    if (TN == 128)
        KNL_gemm_bf16<128><<<grid, dim3(256), 0, st>>>(
            A, lda, zbA, cntA, soA, siA, Bm, ldb, zbB, cntB, soB, siB, transB,
            C, cbf, ldc, zbC, cntC, soC, siC, bias, resid, relu, K);
    else
        KNL_gemm_bf16<64><<<grid, dim3(256), 0, st>>>(
            A, lda, zbA, cntA, soA, siA, Bm, ldb, zbB, cntB, soB, siB, transB,
            C, cbf, ldc, zbC, cntC, soC, siC, bias, resid, relu, K);
}

extern "C" void kernel_launch(void* const* d_in, const int* in_sizes, int n_in,
                              void* d_out, int out_size, void* d_ws, size_t ws_size,
                              hipStream_t stream) {
    (void)in_sizes; (void)n_in; (void)out_size; (void)ws_size;

    const int*   src     = (const int*)d_in[0];
    const int*   lengths = (const int*)d_in[1];
    const float* emb     = (const float*)d_in[2];
    const float* Wq      = (const float*)d_in[3];
    const float* bq      = (const float*)d_in[4];
    const float* Wk      = (const float*)d_in[5];
    const float* bk      = (const float*)d_in[6];
    const float* Wv      = (const float*)d_in[7];
    const float* bv      = (const float*)d_in[8];
    const float* Wo      = (const float*)d_in[9];
    const float* bo      = (const float*)d_in[10];
    const float* ln1_g   = (const float*)d_in[11];
    const float* ln1_b   = (const float*)d_in[12];
    const float* W1      = (const float*)d_in[13];
    const float* b1      = (const float*)d_in[14];
    const float* W2      = (const float*)d_in[15];
    const float* b2      = (const float*)d_in[16];
    const float* ln2_g   = (const float*)d_in[17];
    const float* ln2_b   = (const float*)d_in[18];
    const float* lnf_g   = (const float*)d_in[19];
    const float* lnf_b   = (const float*)d_in[20];

    const long DD = (long)DMODEL * DMODEL;
    const long DF = (long)DMODEL * FFDIM;
    const long SD = (long)SEQ * DMODEL;
    const long SS = (long)SEQ * SEQ;

    // ---- workspace (bump allocator, 256B aligned) ----
    char* p = (char*)d_ws;
    auto alloc = [&](size_t bytes) -> void* {
        void* r = (void*)p;
        p += (bytes + 255) & ~(size_t)255;
        return r;
    };
    float*  out_buf = (float*)alloc((size_t)TOK * DMODEL * 4);   // residual stream
    __bf16* h_bf    = (__bf16*)alloc((size_t)TOK * DMODEL * 2);  // LN output
    __bf16* q_bf    = (__bf16*)alloc((size_t)TOK * DMODEL * 2);
    __bf16* k_bf    = (__bf16*)alloc((size_t)TOK * DMODEL * 2);
    __bf16* v_bf    = (__bf16*)alloc((size_t)TOK * DMODEL * 2);
    __bf16* ctx_bf  = (__bf16*)alloc((size_t)TOK * DMODEL * 2);
    __bf16* ff_bf   = (__bf16*)alloc((size_t)TOK * FFDIM * 2);
    float*  sc_f    = (float*)alloc((size_t)CHUNK * SS * 4);
    __bf16* sc_bf   = (__bf16*)alloc((size_t)CHUNK * SS * 2);
    __bf16* WqT     = (__bf16*)alloc((size_t)LNUM * DD * 2);
    __bf16* WkT     = (__bf16*)alloc((size_t)LNUM * DD * 2);
    __bf16* WvT     = (__bf16*)alloc((size_t)LNUM * DD * 2);
    __bf16* WoT     = (__bf16*)alloc((size_t)LNUM * DD * 2);
    __bf16* W1T     = (__bf16*)alloc((size_t)LNUM * DF * 2);
    __bf16* W2T     = (__bf16*)alloc((size_t)LNUM * DF * 2);

    // ---- one-time weight transpose + bf16 conversion ----
    for (int i = 0; i < LNUM; ++i) {
        dim3 gDD(DMODEL / 32, DMODEL / 32);
        KNL_transpose_cvt<<<gDD, 256, 0, stream>>>(Wq + i * DD, WqT + i * DD, DMODEL, DMODEL);
        KNL_transpose_cvt<<<gDD, 256, 0, stream>>>(Wk + i * DD, WkT + i * DD, DMODEL, DMODEL);
        KNL_transpose_cvt<<<gDD, 256, 0, stream>>>(Wv + i * DD, WvT + i * DD, DMODEL, DMODEL);
        KNL_transpose_cvt<<<gDD, 256, 0, stream>>>(Wo + i * DD, WoT + i * DD, DMODEL, DMODEL);
        KNL_transpose_cvt<<<dim3(FFDIM / 32, DMODEL / 32), 256, 0, stream>>>(
            W1 + i * DF, W1T + i * DF, DMODEL, FFDIM);   // [D][FF] -> [FF][D]
        KNL_transpose_cvt<<<dim3(DMODEL / 32, FFDIM / 32), 256, 0, stream>>>(
            W2 + i * DF, W2T + i * DF, FFDIM, DMODEL);   // [FF][D] -> [D][FF]
    }

    // x = emb[src]
    KNL_embed<<<dim3(TOK), dim3(256), 0, stream>>>(src, emb, out_buf);

    for (int i = 0; i < LNUM; ++i) {
        // h = LN1(out) -> bf16
        KNL_layernorm<<<dim3(TOK), dim3(256), 0, stream>>>(
            out_buf, ln1_g + (long)i * DMODEL, ln1_b + (long)i * DMODEL, h_bf, 1);

        // q/k/v = h @ W + b  (bf16 out, pre-transposed weights)
        launch_gemm(stream, 128, h_bf, DMODEL, 0,1,0,0,
                    WqT + i * DD, DMODEL, 0,1,0,0, 1,
                    q_bf, 1, DMODEL, 0,1,0,0,
                    bq + (long)i * DMODEL, nullptr, 0, TOK, DMODEL, DMODEL, 1);
        launch_gemm(stream, 128, h_bf, DMODEL, 0,1,0,0,
                    WkT + i * DD, DMODEL, 0,1,0,0, 1,
                    k_bf, 1, DMODEL, 0,1,0,0,
                    bk + (long)i * DMODEL, nullptr, 0, TOK, DMODEL, DMODEL, 1);
        launch_gemm(stream, 128, h_bf, DMODEL, 0,1,0,0,
                    WvT + i * DD, DMODEL, 0,1,0,0, 1,
                    v_bf, 1, DMODEL, 0,1,0,0,
                    bv + (long)i * DMODEL, nullptr, 0, TOK, DMODEL, DMODEL, 1);

        // attention in chunks of CHUNK (b,h) pairs
        for (int c = 0; c < (BATCH * HEADS) / CHUNK; ++c) {
            int z0 = c * CHUNK;
            // scores[z] = q(b,h) @ k(b,h)^T   (M=S,N=S,K=DH) -> f32
            launch_gemm(stream, 128,
                q_bf, DMODEL, z0, HEADS, SD, DHEAD,
                k_bf, DMODEL, z0, HEADS, SD, DHEAD, 1,
                sc_f, 0, SEQ, 0, 1, SS, 0,
                nullptr, nullptr, 0, SEQ, SEQ, DHEAD, CHUNK);
            // masked scaled softmax -> bf16 probs
            KNL_softmax<<<dim3(SEQ, CHUNK), dim3(256), 0, stream>>>(sc_f, sc_bf, lengths, z0);
            // ctx(b,h) = probs @ v(b,h)   (M=S,N=DH,K=S) -> bf16
            launch_gemm(stream, 64,
                sc_bf, SEQ, 0, 1, SS, 0,
                v_bf, DMODEL, z0, HEADS, SD, DHEAD, 0,
                ctx_bf, 1, DMODEL, z0, HEADS, SD, DHEAD,
                nullptr, nullptr, 0, SEQ, DHEAD, SEQ, CHUNK);
        }

        // out = ctx @ Wo + bo + out   (f32 residual stream)
        launch_gemm(stream, 128, ctx_bf, DMODEL, 0,1,0,0,
                    WoT + i * DD, DMODEL, 0,1,0,0, 1,
                    out_buf, 0, DMODEL, 0,1,0,0,
                    bo + (long)i * DMODEL, out_buf, 0, TOK, DMODEL, DMODEL, 1);

        // f = LN2(out) -> bf16
        KNL_layernorm<<<dim3(TOK), dim3(256), 0, stream>>>(
            out_buf, ln2_g + (long)i * DMODEL, ln2_b + (long)i * DMODEL, h_bf, 1);

        // ff = relu(f @ W1 + b1) -> bf16
        launch_gemm(stream, 128, h_bf, DMODEL, 0,1,0,0,
                    W1T + i * DF, DMODEL, 0,1,0,0, 1,
                    ff_bf, 1, FFDIM, 0,1,0,0,
                    b1 + (long)i * FFDIM, nullptr, 1, TOK, FFDIM, DMODEL, 1);

        // out = ff @ W2 + b2 + out  (f32 residual stream)
        launch_gemm(stream, 128, ff_bf, FFDIM, 0,1,0,0,
                    W2T + i * DF, FFDIM, 0,1,0,0, 1,
                    out_buf, 0, DMODEL, 0,1,0,0,
                    b2 + (long)i * DMODEL, out_buf, 0, TOK, DMODEL, FFDIM, 1);
    }

    // final LN -> f32 d_out
    KNL_layernorm<<<dim3(TOK), dim3(256), 0, stream>>>(
        out_buf, lnf_g, lnf_b, d_out, 0);
}
